// MessageFunction_45303315038658
// MI455X (gfx1250) — matrix operand order repaired and checked by reading, not compiled, not run on hardware
//
#include <hip/hip_runtime.h>

#define BB 16
#define NN 512
#define DD 64
#define EE 8

typedef __attribute__((ext_vector_type(16))) __bf16 v16bf;
typedef __attribute__((ext_vector_type(8)))  float  v8f;

union BFVec { unsigned u[8]; v16bf v; };

// exact parameter types of the async-to-LDS builtins (from hipcc diagnostics):
//   arg0: as(1) pointer to int __vector_size__(8|16)   (global source)
//   arg1: as(3) pointer to same vector type            (LDS destination)
typedef int avec2 __attribute__((vector_size(8)));
typedef int avec4 __attribute__((vector_size(16)));
#define G2(p) ((__attribute__((address_space(1))) avec2*)(uintptr_t)(p))
#define L2(p) ((__attribute__((address_space(3))) avec2*)(uintptr_t)(p))
#define G4(p) ((__attribute__((address_space(1))) avec4*)(uintptr_t)(p))
#define L4(p) ((__attribute__((address_space(3))) avec4*)(uintptr_t)(p))

#if __has_builtin(__builtin_amdgcn_global_load_async_to_lds_b128) && \
    __has_builtin(__builtin_amdgcn_global_load_async_to_lds_b64)
#define USE_ASYNC_LDS 1
#else
#define USE_ASYNC_LDS 0
#endif

__device__ __forceinline__ void wait_async_done() {
#if USE_ASYNC_LDS
#if __has_builtin(__builtin_amdgcn_s_wait_asynccnt)
  __builtin_amdgcn_s_wait_asynccnt(0);
#else
  asm volatile("s_wait_asynccnt 0x0" ::: "memory");
#endif
#endif
}

// round-to-nearest-even f32 -> bf16 (low 16 bits of return)
__device__ __forceinline__ unsigned f2bf1(float f) {
  unsigned u = __builtin_bit_cast(unsigned, f);
  return (u + 0x7FFFu + ((u >> 16) & 1u)) >> 16;
}
__device__ __forceinline__ unsigned packbf(float lo, float hi) {
#if __has_builtin(__builtin_amdgcn_cvt_pk_bf16_f32)
  typedef __attribute__((ext_vector_type(2))) __bf16 v2bf;
  v2bf p = __builtin_amdgcn_cvt_pk_bf16_f32(lo, hi);
  return __builtin_bit_cast(unsigned, p);
#else
  return (f2bf1(lo) & 0xFFFFu) | (f2bf1(hi) << 16);
#endif
}

// A-matrix (16x32 bf16) K index for VGPR v, lane-half h (0:lanes0-15, 1:lanes16-31)
__device__ __forceinline__ int a_kmap(int v, int h) {
  return ((v < 4) ? (2 * v) : (16 + 2 * (v - 4))) + (h ? 8 : 0);
}

__global__ __launch_bounds__(256) void ggnn_fused_kernel(
    const float* __restrict__ hsrc,   // [B,N,D] f32
    const int*   __restrict__ adj,    // [B,N,N] int32 in [0,E)
    const float* __restrict__ Win,    // [E,D,D]
    const float* __restrict__ Wout,   // [E,D,D]
    const float* __restrict__ bias,   // [2D]
    float*       __restrict__ out)    // [B,N,2D]
{
  __shared__ int            s_adj[16 * 32];        // 2 KB  (per-dir adj tile, 16x32 or 32x16)
  __shared__ float          s_hraw[32 * 64];       // 8 KB  (h chunk, f32)
  __shared__ unsigned       s_bpack[4 * 32 * 8];   // 4 KB  ([nt][lane][v] B-layout bf16 pairs)
  __shared__ unsigned short s_aggb[8 * 16 * 64];   // 16 KB (per-e agg, bf16 row-major [e*16+i][k])
  __shared__ float          s_red[4 * 16 * 64];    // 16 KB (reduction tree regions)

  const int t    = threadIdx.x;
  const int lane = t & 31;
  const int w    = t >> 5;           // wave id == edge class e
  const int m    = lane & 15;
  const int lh   = (lane >> 4) & 1;  // lane-half
  const int blk  = blockIdx.x;
  const int b    = blk >> 5;         // 32 node-tiles per batch
  const int i0   = (blk & 31) << 4;

#pragma unroll
  for (int dir = 0; dir < 2; ++dir) {
    const v8f z = {0.f, 0.f, 0.f, 0.f, 0.f, 0.f, 0.f, 0.f};
    v8f acc[4] = {z, z, z, z};

    // ---------------- Phase 1: one-hot routing via WMMA ----------------
    for (int jc = 0; jc < 16; ++jc) {
      const int j0 = jc << 5;
      __syncthreads();  // previous readers of s_adj/s_hraw/s_bpack are done

      // ---- stage adj tile for this direction + h chunk, direct to LDS (ASYNCcnt path)
      if (dir == 0) {                  // rows: adj[b, i0+il, j0 : j0+32]
        const int il = t >> 4, jj = (t & 15) << 1;
        const int* gsrc = adj + ((size_t)(b * NN + i0 + il) * NN + j0 + jj);
        int* ldst = &s_adj[il * 32 + jj];
#if USE_ASYNC_LDS
        __builtin_amdgcn_global_load_async_to_lds_b64(G2(gsrc), L2(ldst), 0, 0);
#else
        const int2 aa = *(const int2*)gsrc;
        ldst[0] = aa.x; ldst[1] = aa.y;
#endif
      } else {                         // column stripe: adj[b, j0+jl, i0 : i0+16]
        const int jl = t >> 3, ii = (t & 7) << 1;
        const int* gsrc = adj + ((size_t)(b * NN + j0 + jl) * NN + i0 + ii);
        int* ldst = &s_adj[jl * 16 + ii];
#if USE_ASYNC_LDS
        __builtin_amdgcn_global_load_async_to_lds_b64(G2(gsrc), L2(ldst), 0, 0);
#else
        const int2 aa = *(const int2*)gsrc;
        ldst[0] = aa.x; ldst[1] = aa.y;
#endif
      }
      // h chunk (32 rows x 64 f32): 512 float4s, 2 per thread
      {
        int idx = t * 2;
#pragma unroll
        for (int q = 0; q < 2; ++q, ++idx) {
          const int jr = idx >> 4, kc = (idx & 15) << 2;
          const float* gsrc = hsrc + ((size_t)(b * NN + j0 + jr) * DD + kc);
          float* ldst = &s_hraw[jr * 64 + kc];
#if USE_ASYNC_LDS
          __builtin_amdgcn_global_load_async_to_lds_b128(G4(gsrc), L4(ldst), 0, 0);
#else
          *(float4*)ldst = *(const float4*)gsrc;
#endif
        }
      }
      wait_async_done();
      __syncthreads();

      // pack h chunk into WMMA B-matrix register layout (bf16 pairs), shared by all 8 waves
      {
        int idx = t * 4;  // 1024 packed dwords total
#pragma unroll
        for (int q = 0; q < 4; ++q, ++idx) {
          const int nt = idx >> 8, rem = idx & 255, l = rem >> 3, v = rem & 7;
          const int jloc = a_kmap(v, (l >= 16) ? 1 : 0);
          const int kk   = nt * 16 + (l & 15);
          s_bpack[(nt * 32 + l) * 8 + v] =
              packbf(s_hraw[jloc * 64 + kk], s_hraw[(jloc + 1) * 64 + kk]);
        }
      }
      __syncthreads();

      // build one-hot A operand for class e=w straight from LDS adj tile
      BFVec A;
#pragma unroll
      for (int v = 0; v < 8; ++v) {
        const int kk = a_kmap(v, lh);
        int a0, a1;
        if (dir == 0) { a0 = s_adj[m * 32 + kk];  a1 = s_adj[m * 32 + kk + 1]; }
        else          { a0 = s_adj[kk * 16 + m];  a1 = s_adj[(kk + 1) * 16 + m]; }
        A.u[v] = ((a0 == w) ? 0x3F80u : 0u) | ((a1 == w) ? 0x3F800000u : 0u);
      }
#pragma unroll
      for (int nt = 0; nt < 4; ++nt) {
        BFVec Bv;
        const uint4 qlo = *(const uint4*)&s_bpack[(nt * 32 + lane) * 8 + 0];
        const uint4 qhi = *(const uint4*)&s_bpack[(nt * 32 + lane) * 8 + 4];
        Bv.u[0] = qlo.x; Bv.u[1] = qlo.y; Bv.u[2] = qlo.z; Bv.u[3] = qlo.w;
        Bv.u[4] = qhi.x; Bv.u[5] = qhi.y; Bv.u[6] = qhi.z; Bv.u[7] = qhi.w;
        acc[nt] = __builtin_amdgcn_wmma_f32_16x16x32_bf16(
            false, A.v, false, Bv.v, (short)0, acc[nt], false, false);
      }
    }

    // ---------------- Phase 2: class projection agg_e @ W_e^T ----------------
    // dump f32 accumulators as bf16 row-major [e*16+i][k]
#pragma unroll
    for (int nt = 0; nt < 4; ++nt)
#pragma unroll
      for (int r = 0; r < 8; ++r) {
        const int M = r + (lh ? 8 : 0);
        s_aggb[(w * 16 + M) * 64 + nt * 16 + m] = (unsigned short)f2bf1(acc[nt][r]);
      }
    __syncthreads();

    const float* W = dir ? Wout : Win;
    v8f acc2[4] = {z, z, z, z};
#pragma unroll
    for (int kh = 0; kh < 2; ++kh) {
      BFVec A2;
#pragma unroll
      for (int v = 0; v < 8; ++v) {
        const int kk = a_kmap(v, lh) + kh * 32;
        A2.u[v] = *(const unsigned*)&s_aggb[(w * 16 + m) * 64 + kk];
      }
#pragma unroll
      for (int nt = 0; nt < 4; ++nt) {
        BFVec Bw;
#pragma unroll
        for (int v = 0; v < 8; ++v) {
          const int kk = a_kmap(v, lh) + kh * 32;
          const int d  = nt * 16 + m;
          const float2 wv = *(const float2*)(W + ((size_t)(w * DD + d) * DD + kk));
          Bw.u[v] = packbf(wv.x, wv.y);
        }
        acc2[nt] = __builtin_amdgcn_wmma_f32_16x16x32_bf16(
            false, A2.v, false, Bw.v, (short)0, acc2[nt], false, false);
      }
    }

    // deterministic tree reduction over the 8 class partials
    if (w >= 4) {
#pragma unroll
      for (int nt = 0; nt < 4; ++nt)
#pragma unroll
        for (int r = 0; r < 8; ++r)
          s_red[(((w - 4) * 16) + r + (lh ? 8 : 0)) * 64 + nt * 16 + m] = acc2[nt][r];
    }
    __syncthreads();
    if (w < 4) {
#pragma unroll
      for (int nt = 0; nt < 4; ++nt)
#pragma unroll
        for (int r = 0; r < 8; ++r)
          acc2[nt][r] += s_red[((w * 16) + r + (lh ? 8 : 0)) * 64 + nt * 16 + m];
    }
    __syncthreads();
    if (w == 2 || w == 3) {
#pragma unroll
      for (int nt = 0; nt < 4; ++nt)
#pragma unroll
        for (int r = 0; r < 8; ++r)
          s_red[(((w - 2) * 16) + r + (lh ? 8 : 0)) * 64 + nt * 16 + m] = acc2[nt][r];
    }
    __syncthreads();
    if (w < 2) {
#pragma unroll
      for (int nt = 0; nt < 4; ++nt)
#pragma unroll
        for (int r = 0; r < 8; ++r)
          acc2[nt][r] += s_red[((w * 16) + r + (lh ? 8 : 0)) * 64 + nt * 16 + m];
    }
    __syncthreads();
    if (w == 1) {
#pragma unroll
      for (int nt = 0; nt < 4; ++nt)
#pragma unroll
        for (int r = 0; r < 8; ++r)
          s_red[(r + (lh ? 8 : 0)) * 64 + nt * 16 + m] = acc2[nt][r];
    }
    __syncthreads();
    if (w == 0) {
#pragma unroll
      for (int nt = 0; nt < 4; ++nt) {
        const float bv = bias[dir * DD + nt * 16 + m];
#pragma unroll
        for (int r = 0; r < 8; ++r) {
          const int M = r + (lh ? 8 : 0);
          const float val = acc2[nt][r] + s_red[M * 64 + nt * 16 + m] + bv;
          out[((size_t)(b * NN + i0 + M)) * (2 * DD) + dir * DD + nt * 16 + m] = val;
        }
      }
    }
    __syncthreads();  // LDS reused by next direction
  }
}

extern "C" void kernel_launch(void* const* d_in, const int* in_sizes, int n_in,
                              void* d_out, int out_size, void* d_ws, size_t ws_size,
                              hipStream_t stream) {
  (void)in_sizes; (void)n_in; (void)d_ws; (void)ws_size; (void)out_size;
  const float* node_state = (const float*)d_in[0];
  const int*   adj_mat    = (const int*)  d_in[1];
  const float* matrix_in  = (const float*)d_in[2];
  const float* matrix_out = (const float*)d_in[3];
  const float* bias       = (const float*)d_in[4];
  float* out = (float*)d_out;

  dim3 grid(BB * (NN / 16));  // 512 blocks: (batch, 16-node tile)
  dim3 block(256);            // 8 wave32 = 8 edge classes
  ggnn_fused_kernel<<<grid, block, 0, stream>>>(node_state, adj_mat, matrix_in,
                                                matrix_out, bias, out);
}
